// UserModel_44220983279646
// MI455X (gfx1250) — matrix operand order repaired and checked by reading, not compiled, not run on hardware
//
#include <hip/hip_runtime.h>
#include <hip/hip_bf16.h>

// B=64, S=200, DV=128, NC2=4096, ND=10000
#define BB 64
#define SS 200
#define DVV 128
#define NC2V 4096

typedef __attribute__((ext_vector_type(16))) _Float16 v16h;
typedef __attribute__((ext_vector_type(8)))  float    v8f;
typedef __attribute__((ext_vector_type(4)))  float    v4f;

__device__ __forceinline__ v8f wmma32(v16h a, v16h b, v8f c) {
    return __builtin_amdgcn_wmma_f32_16x16x32_f16(false, a, false, b, (short)0, c, false, false);
}

// A tile (16x32, f16) from row-major f16 memory. Lane l holds row (l&15);
// lanes 0-15 carry K 0-7 & 16-23, lanes 16-31 carry K 8-15 & 24-31.
__device__ __forceinline__ v16h load_a_h(const _Float16* base, int stride, int row0, int k0, int lane) {
    int r  = row0 + (lane & 15);
    int kb = (lane < 16) ? 0 : 8;
    const _Float16* p = base + (size_t)r * stride + k0;
    v16h a;
#pragma unroll
    for (int j = 0; j < 8; ++j) { a[j] = p[kb + j]; a[8 + j] = p[16 + kb + j]; }
    return a;
}

// A tile built from f32 memory (LDS or global) with conversion.
__device__ __forceinline__ v16h load_a_f(const float* base, int stride, int row0, int k0, int lane) {
    int r  = row0 + (lane & 15);
    int kb = (lane < 16) ? 0 : 8;
    const float* p = base + (size_t)r * stride + k0;
    v16h a;
#pragma unroll
    for (int j = 0; j < 8; ++j) { a[j] = (_Float16)p[kb + j]; a[8 + j] = (_Float16)p[16 + kb + j]; }
    return a;
}

// B tile (32x16) where B[k][n] = W[n][k], W row-major [N][K] f16.
// Lane l holds column (l&15); lanes 0-15 K 0-7 & 16-23, lanes 16-31 K 8-15 & 24-31.
__device__ __forceinline__ v16h load_b_wt(const _Float16* base, int kstride, int n0, int k0, int lane) {
    int n  = n0 + (lane & 15);
    int kb = (lane < 16) ? 0 : 8;
    const _Float16* p = base + (size_t)n * kstride + k0 + kb;
    v16h b;
#pragma unroll
    for (int j = 0; j < 8; ++j) { b[j] = p[j]; b[8 + j] = p[16 + j]; }
    return b;
}

__device__ __forceinline__ float sigm(float x) { return 1.f / (1.f + __expf(-x)); }

// ---------------- elementwise helpers ----------------
__global__ void k_cvt(const float* __restrict__ src, _Float16* __restrict__ dst, int n) {
    int i = blockIdx.x * 256 + threadIdx.x;
    if (i < n) dst[i] = (_Float16)src[i];
}

__global__ void k_zero(float* p, int n) {
    int i = blockIdx.x * 256 + threadIdx.x;
    if (i < n) p[i] = 0.f;
}

// v_d_seq, v_r_seq (f32) and x (f16, concat) — one block per (b,s)
__global__ void k_embed(const int* __restrict__ d_seq, const int* __restrict__ r_seq,
                        const float* __restrict__ D_emb, const float* __restrict__ v_d,
                        const float* __restrict__ R_emb,
                        float* __restrict__ vdws, float* __restrict__ vrws,
                        _Float16* __restrict__ xf16) {
    int m = blockIdx.x;            // b*S+s
    int j = threadIdx.x;           // 0..127
    float gamma = D_emb[d_seq[m]];
    int r = r_seq[m];
    float vd = gamma * v_d[j];
    float vr = R_emb[r * DVV + j];
    vdws[(size_t)m * DVV + j] = vd;
    vrws[(size_t)m * DVV + j] = vr;
    xf16[(size_t)m * 256 + j]       = (_Float16)vd;
    xf16[(size_t)m * 256 + 128 + j] = (_Float16)vr;
}

// xW = x @ W_ih^T + b_ih   (M=12800, N=384, K=256)
__global__ void k_xw(const _Float16* __restrict__ xf16, const _Float16* __restrict__ Wih,
                     const float* __restrict__ b_ih, float* __restrict__ xW) {
    int lane = threadIdx.x & 31, wave = threadIdx.x >> 5;
    int mtile = blockIdx.x;
    int ntile = blockIdx.y * 8 + wave;   // 0..23
    v8f acc = {};
#pragma unroll
    for (int ks = 0; ks < 8; ++ks) {
        v16h a = load_a_h(xf16, 256, mtile * 16, ks * 32, lane);
        v16h b = load_b_wt(Wih, 256, ntile * 16, ks * 32, lane);
        acc = wmma32(a, b, acc);
    }
    int n = ntile * 16 + (lane & 15);
    float bias = b_ih[n];
    int mbase = mtile * 16 + ((lane < 16) ? 0 : 8);
#pragma unroll
    for (int r2 = 0; r2 < 8; ++r2)
        xW[(size_t)(mbase + r2) * 384 + n] = acc[r2] + bias;
}

// GRU scan: single workgroup, 32 waves; wave owns one 16x16 (b,j) block and
// computes its r/z/n tiles itself -> gate math stays register-local.
__global__ void __launch_bounds__(1024) k_gru(const float* __restrict__ xW,
                                              const _Float16* __restrict__ Whh,
                                              const float* __restrict__ b_hh,
                                              float* __restrict__ h_seq) {
    __shared__ float hL[BB * DVV];     // 32 KB
    int tid = threadIdx.x;
    int lane = tid & 31, wave = tid >> 5;
    int mtile = wave >> 3;             // 0..3
    int ntile = wave & 7;              // 0..7
    for (int i = tid; i < BB * DVV; i += 1024) hL[i] = 0.f;
    __syncthreads();
    int col  = lane & 15;
    int jabs = ntile * 16 + col;
    float bhr = b_hh[jabs], bhz = b_hh[128 + jabs], bhn = b_hh[256 + jabs];
    for (int s = 0; s < SS; ++s) {
        v8f ar = {}, az = {}, an = {};
#pragma unroll
        for (int ks = 0; ks < 4; ++ks) {
            v16h a  = load_a_f(hL, DVV, mtile * 16, ks * 32, lane);
            v16h br = load_b_wt(Whh, DVV, 0 * 128 + ntile * 16, ks * 32, lane);
            v16h bz = load_b_wt(Whh, DVV, 1 * 128 + ntile * 16, ks * 32, lane);
            v16h bn = load_b_wt(Whh, DVV, 2 * 128 + ntile * 16, ks * 32, lane);
            ar = wmma32(a, br, ar);
            az = wmma32(a, bz, az);
            an = wmma32(a, bn, an);
        }
        __syncthreads();               // everyone done reading hL
        int mbase = mtile * 16 + ((lane < 16) ? 0 : 8);
#pragma unroll
        for (int r2 = 0; r2 < 8; ++r2) {
            int b = mbase + r2;
            const float* xrow = &xW[(size_t)(b * SS + s) * 384];
            float xr = xrow[jabs], xz = xrow[128 + jabs], xn = xrow[256 + jabs];
            float hold = hL[b * DVV + jabs];
            float rg = sigm(xr + ar[r2] + bhr);
            float zg = sigm(xz + az[r2] + bhz);
            float ng = tanhf(xn + rg * (an[r2] + bhn));
            float hnew = (1.f - zg) * ng + zg * hold;
            hL[b * DVV + jabs] = hnew;
            h_seq[(size_t)(b * SS + s) * DVV + jabs] = hnew;
        }
        __syncthreads();
    }
}

// alpha = relu(h @ W1^T + b1) @ W2^T + b2, fused per 16-row panel
__global__ void k_alpha(const float* __restrict__ h_seq, const _Float16* __restrict__ W1f,
                        const float* __restrict__ b1, const float* __restrict__ W2,
                        const float* __restrict__ b2, float* __restrict__ alpha) {
    __shared__ float rl[16 * DVV];     // 8 KB
    int tid = threadIdx.x, lane = tid & 31, wave = tid >> 5;
    int mtile = blockIdx.x, ntile = wave;
    int col = lane & 15;
    v8f acc = {};
#pragma unroll
    for (int ks = 0; ks < 4; ++ks) {
        v16h a = load_a_f(h_seq, DVV, mtile * 16, ks * 32, lane);
        v16h b = load_b_wt(W1f, DVV, ntile * 16, ks * 32, lane);
        acc = wmma32(a, b, acc);
    }
    int jabs = ntile * 16 + col;
    float bb = b1[jabs];
    int mbase = (lane < 16) ? 0 : 8;
#pragma unroll
    for (int r2 = 0; r2 < 8; ++r2) {
        float v = acc[r2] + bb;
        rl[(mbase + r2) * DVV + jabs] = v > 0.f ? v : 0.f;
    }
    __syncthreads();
    if (tid < 16) {
        float s = b2[0];
        for (int j = 0; j < DVV; ++j) s += rl[tid * DVV + j] * W2[j];
        alpha[mtile * 16 + tid] = s;
    }
}

// C2 scan: single workgroup computes the 200x64 scalar updates (WMMA feat GEMM),
// keeps the 64x4096 state in global workspace (only 64 R/W per step).
__global__ void __launch_bounds__(256) k_c2scan(const int* __restrict__ c2_seq,
                                                const float* __restrict__ vdws,
                                                const float* __restrict__ vrws,
                                                const float* __restrict__ v_c2,
                                                const _Float16* __restrict__ W3f,
                                                const float* __restrict__ b3,
                                                const float* __restrict__ W4,
                                                const float* __restrict__ b4,
                                                float* __restrict__ C2ws,
                                                float* __restrict__ newvals) {
    __shared__ _Float16 featL[BB * 384];   // 48 KB
    __shared__ float betaL[BB];
    __shared__ float accL[BB];
    int tid = threadIdx.x, lane = tid & 31, wave = tid >> 5;
    int col = lane & 15;
    for (int t = 0; t < SS; ++t) {
        if (tid < 64)       betaL[tid] = C2ws[(size_t)tid * NC2V + c2_seq[tid * SS + t]];
        else if (tid < 128) accL[tid - 64] = 0.f;
        __syncthreads();
        for (int idx = tid; idx < BB * 384; idx += 256) {
            int b = idx / 384, c = idx % 384;
            float v;
            if (c < 128)      v = betaL[b] * v_c2[c];
            else if (c < 256) v = vdws[(size_t)(b * SS + t) * DVV + (c - 128)];
            else              v = vrws[(size_t)(b * SS + t) * DVV + (c - 256)];
            featL[idx] = (_Float16)v;
        }
        __syncthreads();
#pragma unroll
        for (int tt = 0; tt < 4; ++tt) {
            int tile = wave * 4 + tt;       // 32 tiles total
            int mtile = tile >> 3, ntile = tile & 7;
            v8f acc = {};
#pragma unroll
            for (int ks = 0; ks < 12; ++ks) {
                v16h a = load_a_h(featL, 384, mtile * 16, ks * 32, lane);
                v16h b = load_b_wt(W3f, 384, ntile * 16, ks * 32, lane);
                acc = wmma32(a, b, acc);
            }
            int jabs = ntile * 16 + col;
            float w4j = W4[jabs], b3j = b3[jabs];
            int mbase = mtile * 16 + ((lane < 16) ? 0 : 8);
#pragma unroll
            for (int r2 = 0; r2 < 8; ++r2) {
                float v = acc[r2] + b3j;
                v = v > 0.f ? v : 0.f;
                atomicAdd(&accL[mbase + r2], v * w4j);  // ds_add_f32
            }
        }
        __syncthreads();
        if (tid < 64) {
            float nv = accL[tid] + b4[0];
            newvals[t * 64 + tid] = nv;
            C2ws[(size_t)tid * NC2V + c2_seq[tid * SS + t]] = nv;
        }
        __threadfence();
        __syncthreads();
    }
}

// Reconstruct C2_seq snapshots: the 209 MB write-once stream. Each thread owns
// 4 contiguous cols -> one b128 non-temporal store per step (don't evict the
// 192 MB L2 working set with data that is never re-read).
__global__ void k_c2emit(const int* __restrict__ c2_seq, const float* __restrict__ newvals,
                         float* __restrict__ C2_out) {
    __shared__ int   idxL[SS];
    __shared__ float valL[SS];
    int b = blockIdx.y;
    int tid = threadIdx.x;
    if (tid < SS) { idxL[tid] = c2_seq[b * SS + tid]; valL[tid] = newvals[tid * 64 + b]; }
    __syncthreads();
    int col0 = blockIdx.x * 1024 + tid * 4;          // 4 cols per thread
    v4f cur = {0.f, 0.f, 0.f, 0.f};
    v4f* out = (v4f*)(C2_out + (size_t)b * SS * NC2V + col0);
    for (int s = 0; s < SS; ++s) {
        int d = idxL[s] - col0;                       // 0..3 if this thread owns it
        if ((unsigned)d < 4u) {
            float v = valL[s];
            if (d == 0) cur[0] = v;
            else if (d == 1) cur[1] = v;
            else if (d == 2) cur[2] = v;
            else cur[3] = v;
        }
        __builtin_nontemporal_store(cur, out);        // global_store_b128 ... NT
        out += NC2V / 4;
    }
}

extern "C" void kernel_launch(void* const* d_in, const int* in_sizes, int n_in,
                              void* d_out, int out_size, void* d_ws, size_t ws_size,
                              hipStream_t stream) {
    const int*   c2_seq = (const int*)d_in[1];
    const int*   d_seq  = (const int*)d_in[3];
    const int*   r_seq  = (const int*)d_in[4];
    const float* D_emb  = (const float*)d_in[5];
    const float* v_d    = (const float*)d_in[6];
    const float* v_c2   = (const float*)d_in[7];
    const float* R_emb  = (const float*)d_in[8];
    const float* W_ih   = (const float*)d_in[9];
    const float* W_hh   = (const float*)d_in[10];
    const float* b_ih   = (const float*)d_in[11];
    const float* b_hh   = (const float*)d_in[12];
    const float* W1     = (const float*)d_in[13];
    const float* b1     = (const float*)d_in[14];
    const float* W2     = (const float*)d_in[15];
    const float* b2     = (const float*)d_in[16];
    const float* W3     = (const float*)d_in[17];
    const float* b3     = (const float*)d_in[18];
    const float* W4     = (const float*)d_in[19];
    const float* b4     = (const float*)d_in[20];

    char* ws = (char*)d_ws;
    size_t off = 0;
    auto alloc = [&](size_t bytes) -> void* {
        void* p = ws + off;
        off = (off + bytes + 255) & ~(size_t)255;
        return p;
    };
    _Float16* xf16 = (_Float16*)alloc((size_t)12800 * 256 * 2);
    _Float16* Wihf = (_Float16*)alloc((size_t)384 * 256 * 2);
    _Float16* Whhf = (_Float16*)alloc((size_t)384 * 128 * 2);
    _Float16* W1f  = (_Float16*)alloc((size_t)128 * 128 * 2);
    _Float16* W3f  = (_Float16*)alloc((size_t)128 * 384 * 2);
    float* xW      = (float*)alloc((size_t)12800 * 384 * 4);
    float* vdws    = (float*)alloc((size_t)12800 * 128 * 4);
    float* vrws    = (float*)alloc((size_t)12800 * 128 * 4);
    float* C2ws    = (float*)alloc((size_t)64 * 4096 * 4);
    float* newvals = (float*)alloc((size_t)200 * 64 * 4);

    float* alpha_out = (float*)d_out;                        // [B,S]
    float* h_out     = alpha_out + 12800;                    // [B,S,DV]
    float* C2_out    = h_out + (size_t)12800 * 128;          // [B,S,NC2]

    k_cvt<<<(98304 + 255) / 256, 256, 0, stream>>>(W_ih, Wihf, 98304);
    k_cvt<<<(49152 + 255) / 256, 256, 0, stream>>>(W_hh, Whhf, 49152);
    k_cvt<<<(16384 + 255) / 256, 256, 0, stream>>>(W1,  W1f, 16384);
    k_cvt<<<(49152 + 255) / 256, 256, 0, stream>>>(W3,  W3f, 49152);

    k_embed<<<12800, 128, 0, stream>>>(d_seq, r_seq, D_emb, v_d, R_emb, vdws, vrws, xf16);
    k_xw<<<dim3(800, 3), 256, 0, stream>>>(xf16, Wihf, b_ih, xW);
    k_gru<<<1, 1024, 0, stream>>>(xW, Whhf, b_hh, h_out);
    k_alpha<<<800, 256, 0, stream>>>(h_out, W1f, b1, W2, b2, alpha_out);

    k_zero<<<(262144 + 255) / 256, 256, 0, stream>>>(C2ws, 262144);
    k_c2scan<<<1, 256, 0, stream>>>(c2_seq, vdws, vrws, v_c2, W3f, b3, W4, b4, C2ws, newvals);
    k_c2emit<<<dim3(4, 64), 256, 0, stream>>>(c2_seq, newvals, C2_out);
}
